// SparseAttention_75565654605909
// MI455X (gfx1250) — matrix operand order repaired and checked by reading, not compiled, not run on hardware
//
#include <hip/hip_runtime.h>
#include <stdint.h>

// CDNA5 / gfx1250 flash attention with 64x64 block-mask tile skipping.
// wave32, v_wmma_f32_16x16x32_f16, 4 waves per workgroup = 64 query rows.
// - K/V tiles cooperatively staged f32->f16 in LDS (conflict-free 144B rows)
// - softmax row-sum computed by WMMA via an all-ones V column (5th acc tile)
// - row-max reduction done with DPP butterflies (no LDS bpermute traffic)
// - exp2-domain softmax: log2(e) folded into the Q scale -> bare v_exp_f32

typedef __attribute__((ext_vector_type(16))) _Float16 v16h;
typedef __attribute__((ext_vector_type(4)))  _Float16 v4h;
typedef __attribute__((ext_vector_type(8)))  float    v8f;

#define S_LEN   2048
#define D_DIM   64
#define NHEADS  16
#define BLK     64
#define STR     72   // padded LDS row stride (halves): 144 B -> bank-conflict-free

// ---- operand packing (per CDNA5 ISA 7.12.2 VGPR layouts) -------------------
// A (16x32 f16): lanes 0-15 hold K {0..7,16..23}; lanes 16-31 hold {8..15,24..31}
__device__ __forceinline__ v16h pack_a_f32(const float* p, float mul) {
  v16h r;
#pragma unroll
  for (int i = 0; i < 8; ++i) r[i]     = (_Float16)(p[i] * mul);
#pragma unroll
  for (int i = 0; i < 8; ++i) r[i + 8] = (_Float16)(p[16 + i] * mul);
  return r;
}
__device__ __forceinline__ v16h pack_a_h(const _Float16* p) {
  v16h r;
#pragma unroll
  for (int i = 0; i < 8; ++i) r[i]     = p[i];
#pragma unroll
  for (int i = 0; i < 8; ++i) r[i + 8] = p[16 + i];
  return r;
}
// B (32x16 f16): same K order as A: lanes 0-15 hold K {0..7,16..23}; lanes 16-31 hold {8..15,24..31}
__device__ __forceinline__ v16h pack_b_h(const _Float16* p) {
  v16h r;
#pragma unroll
  for (int i = 0; i < 16; ++i) r[i] = p[(i & 7) + ((i >> 3) << 4)];
  return r;
}

// ---- DPP 16-lane max butterfly (stays inside wave32 16-lane halves) --------
template <int CTRL>
__device__ __forceinline__ float dpp_mov(float x) {
  int r = __builtin_amdgcn_update_dpp(0, __float_as_int(x), CTRL, 0xF, 0xF, true);
  return __int_as_float(r);
}
__device__ __forceinline__ float red_max16(float x) {
  x = fmaxf(x, dpp_mov<0xB1>(x));   // quad_perm [1,0,3,2]  : xor 1
  x = fmaxf(x, dpp_mov<0x4E>(x));   // quad_perm [2,3,0,1]  : xor 2
  x = fmaxf(x, dpp_mov<0x141>(x));  // row_half_mirror      : xor 7 (within 8)
  x = fmaxf(x, dpp_mov<0x140>(x));  // row_mirror           : xor 15 (within 16)
  return x;
}

__global__ __launch_bounds__(128, 1)
void SparseAttention_75565654605909_kernel(const float* __restrict__ q,
                                           const float* __restrict__ k,
                                           const float* __restrict__ v,
                                           const uint8_t* __restrict__ mask,
                                           float* __restrict__ out) {
  __shared__ _Float16 Kt[BLK * STR];            // f16 K tile, row-major [key][d]
  __shared__ _Float16 Vt[80 * STR];             // f16 V^T tile [d][key]; rows 64..79
                                                // form the ones-column sum tile
  __shared__ _Float16 Plds[4 * 16 * STR];       // per-wave P scratch [row][key]

  const int tid   = threadIdx.x;
  const int wave  = tid >> 5;
  const int lane  = tid & 31;
  const int laneN = lane & 15;
  const int hi    = lane >> 4;
  const int s0a   = hi * 8;    // A-layout segment base
  const int s0b   = hi * 8;    // B-layout segment base

  const int qblk = blockIdx.x;          // 0..31
  const int h    = blockIdx.y;          // 0..15
  const int qb0  = qblk * BLK;

  const float*   qg   = q + (size_t)h * S_LEN * D_DIM;
  const float*   kg   = k + (size_t)h * S_LEN * D_DIM;
  const float*   vg   = v + (size_t)h * S_LEN * D_DIM;
  const uint8_t* mrow = mask + ((size_t)h * S_LEN + qb0) * S_LEN;

  v8f zero8;
#pragma unroll
  for (int i = 0; i < 8; ++i) zero8[i] = 0.0f;

  // Vt rows 64..79: row 64 = ones (sum column), rest zero. Written once.
  for (int i = tid; i < 16 * STR; i += 128) {
    int rr = i / STR, cc = i - rr * STR;
    Vt[(64 + rr) * STR + cc] = (rr == 0 && cc < BLK) ? (_Float16)1.0f
                                                     : (_Float16)0.0f;
  }
  __syncthreads();

  // Persistent Q operands: pre-scaled by log2(e)/sqrt(D) -> exp2-domain logits
  const float qscale = 0.125f * 1.4426950408889634f;
  const float* qp = qg + (size_t)(qb0 + wave * 16 + laneN) * D_DIM;
  const v16h aQ0 = pack_a_f32(qp + s0a,      qscale);   // d = 0..31
  const v16h aQ1 = pack_a_f32(qp + 32 + s0a, qscale);   // d = 32..63

  // Online-softmax state: VGPR r holds row (r + 8*hi), max replicated in lanes.
  // o[0..3] = output numerator tiles, o[4] = row-sum tile (ones column).
  float m_i[8];
  v8f   o[5];
#pragma unroll
  for (int r = 0; r < 8; ++r) m_i[r] = -INFINITY;
#pragma unroll
  for (int t = 0; t < 5; ++t) o[t] = zero8;

  _Float16* Pw = Plds + wave * 16 * STR;

  for (int kt = 0; kt < S_LEN / BLK; ++kt) {
    // mask is constant inside a 64x64 tile; uniform across the workgroup
    if (!mrow[kt * BLK]) continue;

    __syncthreads();  // prior-iteration Kt/Vt readers done

    // ---- cooperative K stage: global f32 [key][d] -> LDS f16 Kt[key][d] ----
    const float* kbase = kg + (size_t)kt * BLK * D_DIM;
#pragma unroll
    for (int it = 0; it < 8; ++it) {
      int idx = it * 128 + tid;              // 1024 float4 groups
      int kk  = idx >> 4;
      int d4  = (idx & 15) * 4;
      float4 val = ((const float4*)kbase)[idx];
      v4h hv;
      hv[0] = (_Float16)val.x; hv[1] = (_Float16)val.y;
      hv[2] = (_Float16)val.z; hv[3] = (_Float16)val.w;
      *(v4h*)&Kt[kk * STR + d4] = hv;        // ds_store_b64, 8B aligned
    }

    // ---- cooperative V stage: global f32 [key][d] -> LDS f16 Vt[d][key] ----
    const float* vbase = vg + (size_t)kt * BLK * D_DIM;
#pragma unroll
    for (int it = 0; it < 2; ++it) {
      int idx = it * 128 + tid;              // 256 quad-groups
      int k4  = (idx >> 4) * 4;              // keys k4..k4+3
      int d4  = (idx & 15) * 4;              // dims d4..d4+3
      float4 r0 = ((const float4*)vbase)[(k4 + 0) * 16 + (d4 >> 2)];
      float4 r1 = ((const float4*)vbase)[(k4 + 1) * 16 + (d4 >> 2)];
      float4 r2 = ((const float4*)vbase)[(k4 + 2) * 16 + (d4 >> 2)];
      float4 r3 = ((const float4*)vbase)[(k4 + 3) * 16 + (d4 >> 2)];
      v4h c0, c1, c2, c3;
      c0[0]=(_Float16)r0.x; c0[1]=(_Float16)r1.x; c0[2]=(_Float16)r2.x; c0[3]=(_Float16)r3.x;
      c1[0]=(_Float16)r0.y; c1[1]=(_Float16)r1.y; c1[2]=(_Float16)r2.y; c1[3]=(_Float16)r3.y;
      c2[0]=(_Float16)r0.z; c2[1]=(_Float16)r1.z; c2[2]=(_Float16)r2.z; c2[3]=(_Float16)r3.z;
      c3[0]=(_Float16)r0.w; c3[1]=(_Float16)r1.w; c3[2]=(_Float16)r2.w; c3[3]=(_Float16)r3.w;
      *(v4h*)&Vt[(d4 + 0) * STR + k4] = c0;
      *(v4h*)&Vt[(d4 + 1) * STR + k4] = c1;
      *(v4h*)&Vt[(d4 + 2) * STR + k4] = c2;
      *(v4h*)&Vt[(d4 + 3) * STR + k4] = c3;
    }

    __syncthreads();  // Kt/Vt fully staged

    // ---- S' = (log2e/sqrt(D)) * Q*K^T : 16 rows x 64 keys (from LDS) -------
    v8f c[4];
#pragma unroll
    for (int nb = 0; nb < 4; ++nb) {
      const _Float16* kp = Kt + (nb * 16 + laneN) * STR;
      v16h b0 = pack_b_h(kp + s0b);          // d = 0..31
      v16h b1 = pack_b_h(kp + 32 + s0b);     // d = 32..63
      v8f cc = zero8;
      cc = __builtin_amdgcn_wmma_f32_16x16x32_f16(false, aQ0, false, b0,
                                                  (short)0, cc, false, false);
      cc = __builtin_amdgcn_wmma_f32_16x16x32_f16(false, aQ1, false, b1,
                                                  (short)0, cc, false, false);
      c[nb] = cc;
    }

    // ---- online softmax: DPP row-max, exp2, alpha-rescale ------------------
    float alpha[8];
#pragma unroll
    for (int r = 0; r < 8; ++r) {
      float mx = fmaxf(fmaxf(c[0][r], c[1][r]), fmaxf(c[2][r], c[3][r]));
      mx = red_max16(mx);                  // uniform across the 16-lane half
      float mn = fmaxf(m_i[r], mx);
      alpha[r] = exp2f(m_i[r] - mn);       // exp2(-inf)=0 on first active tile
      m_i[r]   = mn;
    }
#pragma unroll
    for (int r = 0; r < 8; ++r) {
#pragma unroll
      for (int nb = 0; nb < 4; ++nb) {
        float e = exp2f(c[nb][r] - m_i[r]);
        Pw[(r + 8 * hi) * STR + nb * 16 + laneN] = (_Float16)e;
      }
#pragma unroll
      for (int t = 0; t < 5; ++t) o[t][r] *= alpha[r];
    }

    // ---- [O | l] += P * [V | 1]  (5th tile's column 0 = row sums) ----------
    const _Float16* Prow = Pw + laneN * STR;
    v16h aP0 = pack_a_h(Prow + s0a);        // keys 0..31
    v16h aP1 = pack_a_h(Prow + 32 + s0a);   // keys 32..63
#pragma unroll
    for (int t = 0; t < 5; ++t) {
      const _Float16* vrow = Vt + (t * 16 + laneN) * STR;
      v16h b0 = pack_b_h(vrow + s0b);
      v16h b1 = pack_b_h(vrow + 32 + s0b);
      o[t] = __builtin_amdgcn_wmma_f32_16x16x32_f16(false, aP0, false, b0,
                                                    (short)0, o[t], false, false);
      o[t] = __builtin_amdgcn_wmma_f32_16x16x32_f16(false, aP1, false, b1,
                                                    (short)0, o[t], false, false);
    }
  }

  // ---- normalize and store (fully-masked rows: l==0 -> 0) ------------------
  float* outp = out + (size_t)h * S_LEN * D_DIM;
#pragma unroll
  for (int r = 0; r < 8; ++r) {
    float lsum = __shfl(o[4][r], hi << 4, 32);   // column 0 of the sum tile
    float linv = (lsum > 0.0f) ? (1.0f / lsum) : 0.0f;
    float* op = outp + (size_t)(qb0 + wave * 16 + r + 8 * hi) * D_DIM + laneN;
#pragma unroll
    for (int t = 0; t < 4; ++t) op[t * 16] = o[t][r] * linv;
  }
}

extern "C" void kernel_launch(void* const* d_in, const int* in_sizes, int n_in,
                              void* d_out, int out_size, void* d_ws, size_t ws_size,
                              hipStream_t stream) {
  (void)in_sizes; (void)n_in; (void)out_size; (void)d_ws; (void)ws_size;
  const float*   q    = (const float*)d_in[0];
  const float*   k    = (const float*)d_in[1];
  const float*   v    = (const float*)d_in[2];
  const uint8_t* mask = (const uint8_t*)d_in[3];
  float* out = (float*)d_out;

  dim3 grid(S_LEN / BLK, NHEADS);   // 32 q-blocks x 16 heads
  dim3 block(128);                  // 4 wave32 = 64 query rows
  hipLaunchKernelGGL(SparseAttention_75565654605909_kernel, grid, block, 0, stream,
                     q, k, v, mask, out);
}
